// LocalSelfAttention_6339371729226
// MI455X (gfx1250) — compile-verified, matrix operand order checked by reading, not run on hardware
//
#include <hip/hip_runtime.h>
#include <hip/hip_bf16.h>

// ---------------------------------------------------------------------------
// DeBERTa-style banded local self-attention for MI455X (gfx1250, wave32).
// All matmuls via v_wmma_f32_16x16x32_bf16 (bf16 in, f32 accumulate).
// Round 2: 16x64 tiles/wave in the big GEMMs, transposed position tables and
// V so every WMMA fragment is fed by contiguous b128 loads, global_prefetch.
// ---------------------------------------------------------------------------

typedef __attribute__((ext_vector_type(16))) __bf16 v16bf;
typedef __attribute__((ext_vector_type(8)))  float  v8f;
typedef __attribute__((ext_vector_type(4)))  unsigned int u32x4;

#define S_LEN 2048
#define HD    1024
#define NH    16
#define DH    64
#define WIN   128
#define BAND  272              // 17 tiles of 16 covering 2W+1 = 257 (padded)
#define BANDP 288              // 9 * 32, padded band for WMMA K-chunks
#define FMIN  (-3.4028235e38f)

__device__ __forceinline__ v8f wmma_bf16(v16bf a, v16bf b, v8f c) {
    return __builtin_amdgcn_wmma_f32_16x16x32_bf16(
        /*neg_a=*/false, a, /*neg_b=*/false, b,
        /*c_mod=*/(short)0, c, /*reuse_a=*/false, /*reuse_b=*/false);
}

// Lane's A/B fragment slice for 16x16x32 bf16:
// elements [0..7] = p[0..7], elements [8..15] = p[16..23] (two b128 loads).
// Caller passes p = row + kb + (lane>=16 ? 8 : 0).
__device__ __forceinline__ v16bf load_frag16(const __bf16* p) {
    union { u32x4 q[2]; v16bf v; } u;
    u.q[0] = *(const u32x4*)(p);
    u.q[1] = *(const u32x4*)(p + 16);
    return u.v;
}

// ---------------------------------------------------------------------------
// float32 -> bf16 elementwise convert
// ---------------------------------------------------------------------------
__global__ void cvt_f32_bf16(const float* __restrict__ in,
                             __bf16* __restrict__ out, int n) {
    int i = blockIdx.x * blockDim.x + threadIdx.x;
    if (i < n) out[i] = (__bf16)in[i];
}

// ---------------------------------------------------------------------------
// Position table transpose: (NH, DH, 257) f32 -> (NH, 272, DH) bf16, rows
// 257..271 zero-filled, so pos GEMM B fragments are contiguous b128 loads.
// ---------------------------------------------------------------------------
__global__ void transpose_pos(const float* __restrict__ in,
                              __bf16* __restrict__ out) {
    int tid = blockIdx.x * blockDim.x + threadIdx.x;
    const int total = NH * BAND * DH;
    if (tid >= total) return;
    int h  = tid / (BAND * DH);
    int r  = tid % (BAND * DH);
    int n  = r / DH;
    int kk = r % DH;
    out[tid] = (n < 257) ? (__bf16)in[(size_t)h * DH * 257 + (size_t)kk * 257 + n]
                         : (__bf16)0.0f;
}

// ---------------------------------------------------------------------------
// V transpose: (S, HD) bf16 -> (NH, DH, S) bf16 so ctx-phase B fragments are
// contiguous along the sequence axis.
// ---------------------------------------------------------------------------
__global__ void transpose_v(const __bf16* __restrict__ V,
                            __bf16* __restrict__ Vt) {
    int tid = blockIdx.x * blockDim.x + threadIdx.x;
    const int total = S_LEN * HD;
    if (tid >= total) return;
    int s = tid / HD;
    int c = tid % HD;          // c = h*64 + dd
    Vt[(size_t)c * S_LEN + s] = V[tid];
}

// ---------------------------------------------------------------------------
// C[M,N] = A[M,K] @ B[N,K]^T   (A,B bf16 row-major, lda=ldb=K).
// One wave computes a 16x64 tile (4 accumulators, A fragment reused 4x).
// Output either bf16 (Cb) or f32 with bias (Cf).
// ---------------------------------------------------------------------------
__global__ __launch_bounds__(256) void gemm_bt4(
        const __bf16* __restrict__ A, const __bf16* __restrict__ B,
        int M, int N, int K,
        __bf16* __restrict__ Cb, float* __restrict__ Cf,
        const float* __restrict__ bias) {
    int lane = threadIdx.x & 31;
    int wave = (blockIdx.x * blockDim.x + threadIdx.x) >> 5;
    int ngrp = N >> 6;                     // groups of 64 columns
    int mt = wave / ngrp;
    int ng = wave % ngrp;
    if (mt >= (M >> 4)) return;
    int hf = lane >> 4;
    int ln = lane & 15;

    const __bf16* arow = A + (size_t)(mt * 16 + ln) * K + hf * 8;
    const __bf16* brow0 = B + (size_t)(ng * 64 + 0  + ln) * K + hf * 8;
    const __bf16* brow1 = B + (size_t)(ng * 64 + 16 + ln) * K + hf * 8;
    const __bf16* brow2 = B + (size_t)(ng * 64 + 32 + ln) * K + hf * 8;
    const __bf16* brow3 = B + (size_t)(ng * 64 + 48 + ln) * K + hf * 8;

    v8f acc0 = {}, acc1 = {}, acc2 = {}, acc3 = {};
    for (int kb = 0; kb < K; kb += 32) {
        __builtin_prefetch(arow + kb + 256, 0, 3);   // global_prefetch_b8
        v16bf af = load_frag16(arow + kb);
        acc0 = wmma_bf16(af, load_frag16(brow0 + kb), acc0);
        acc1 = wmma_bf16(af, load_frag16(brow1 + kb), acc1);
        acc2 = wmma_bf16(af, load_frag16(brow2 + kb), acc2);
        acc3 = wmma_bf16(af, load_frag16(brow3 + kb), acc3);
    }

    v8f accs[4] = {acc0, acc1, acc2, acc3};
#pragma unroll
    for (int s = 0; s < 4; ++s) {
        int col = ng * 64 + s * 16 + ln;
        if (Cf) {
            float bv = bias ? bias[col] : 0.0f;
#pragma unroll
            for (int r = 0; r < 8; ++r)
                Cf[(size_t)(mt * 16 + r + hf * 8) * N + col] = accs[s][r] + bv;
        } else {
#pragma unroll
            for (int r = 0; r < 8; ++r)
                Cb[(size_t)(mt * 16 + r + hf * 8) * N + col] = (__bf16)accs[s][r];
        }
    }
}

// ---------------------------------------------------------------------------
// Per-head position GEMM:  C[h, S, 272] = A[:, h*64:+64] @ PT[h]^T
// A (S, HD) bf16 (head slice, row stride HD); PT (NH, 272, DH) bf16 row-major
// over K=64, so both fragments are contiguous b128 loads.
// ---------------------------------------------------------------------------
__global__ __launch_bounds__(256) void pos_gemm(
        const __bf16* __restrict__ A, const __bf16* __restrict__ PT,
        float* __restrict__ C) {
    int lane = threadIdx.x & 31;
    int wave = (blockIdx.x * blockDim.x + threadIdx.x) >> 5;
    int h = blockIdx.y;
    const int NT = BAND / 16;              // 17 col tiles
    int mt = wave / NT;
    int nt = wave % NT;
    if (mt >= (S_LEN >> 4)) return;
    int hf = lane >> 4, ln = lane & 15;

    const __bf16* arow = A + (size_t)(mt * 16 + ln) * HD + h * DH + hf * 8;
    const __bf16* brow = PT + ((size_t)h * BAND + nt * 16 + ln) * DH + hf * 8;

    v8f acc = {};
    acc = wmma_bf16(load_frag16(arow),      load_frag16(brow),      acc);
    acc = wmma_bf16(load_frag16(arow + 32), load_frag16(brow + 32), acc);

    float* Ch = C + (size_t)h * S_LEN * BAND;
    int n = nt * 16 + ln;
#pragma unroll
    for (int r = 0; r < 8; ++r)
        Ch[(size_t)(mt * 16 + r + hf * 8) * BAND + n] = acc[r];
}

// ---------------------------------------------------------------------------
// Banded attention: one wave per (head, 16-row block).
// scores (17 tiles) -> +c2p/p2c gather -> band/mask -> softmax (LDS) -> P@V.
// ---------------------------------------------------------------------------
__global__ __launch_bounds__(32) void attn_band(
        const __bf16* __restrict__ Q, const __bf16* __restrict__ K,
        const __bf16* __restrict__ Vt,     // (NH, DH, S)
        const float* __restrict__ C2P, const float* __restrict__ P2C,
        const unsigned char* __restrict__ mask,
        __bf16* __restrict__ CTX) {
    __shared__ float  sc[16][BANDP];
    __shared__ __bf16 pr[16][BANDP];

    int lane = threadIdx.x;
    int m = blockIdx.x;
    int h = blockIdx.y;
    int row0 = m * 16;
    int colStart = row0 - WIN;             // multiple of 16 (may be < 0)
    int hf = lane >> 4, ln = lane & 15;

    // Q fragments for this row block (K = 64 -> two 32-chunks), loaded once.
    const __bf16* qrow = Q + (size_t)(row0 + ln) * HD + h * DH + hf * 8;
    v16bf q0 = load_frag16(qrow);
    v16bf q1 = load_frag16(qrow + 32);

    const float* c2ph = C2P + (size_t)h * S_LEN * BAND;
    const float* p2ch = P2C + (size_t)h * S_LEN * BAND;

    // ---- score tiles over the band -------------------------------------
    for (int t = 0; t < 17; ++t) {
        int colbase = colStart + t * 16;
        int j = colbase + ln;
        bool jin = (j >= 0) && (j < S_LEN);
        v16bf b0 = {}, b1 = {};
        if (jin) {
            const __bf16* krow = K + (size_t)j * HD + h * DH + hf * 8;
            b0 = load_frag16(krow);
            b1 = load_frag16(krow + 32);
        }
        v8f acc = {};
        acc = wmma_bf16(q0, b0, acc);
        acc = wmma_bf16(q1, b1, acc);

        bool jm = jin && (mask[j] != 0);
#pragma unroll
        for (int r = 0; r < 8; ++r) {
            int i = row0 + r + hf * 8;
            float val;
            if (!jin) {
                val = FMIN;                // nonexistent column
            } else {
                int rel = j - i + WIN;
                rel = rel < 0 ? 0 : (rel > 2 * WIN ? 2 * WIN : rel);
                int rel2 = i - j + WIN;
                rel2 = rel2 < 0 ? 0 : (rel2 > 2 * WIN ? 2 * WIN : rel2);
                val = acc[r] + c2ph[(size_t)i * BAND + rel]
                             + p2ch[(size_t)j * BAND + rel2];
                int di = i - j; if (di < 0) di = -di;
                bool ext = (di <= WIN) && jm && (mask[i] != 0);
                if (!ext) val += FMIN;     // mimic score + finfo.min*(~ext)
            }
            sc[r + hf * 8][t * 16 + ln] = val;
        }
    }
    __syncthreads();

    // ---- softmax over the 272-wide band (one row per lane 0..15) -------
    if (lane < 16) {
        float rmax = FMIN;
        for (int c = 0; c < BAND; ++c) rmax = fmaxf(rmax, sc[lane][c]);
        float rsum = 0.0f;
        for (int c = 0; c < BAND; ++c) {
            float e = __expf(sc[lane][c] - rmax);
            sc[lane][c] = e;
            rsum += e;
        }
        float inv = 1.0f / rsum;
        for (int c = 0; c < BAND; ++c) pr[lane][c] = (__bf16)(sc[lane][c] * inv);
        for (int c = BAND; c < BANDP; ++c) pr[lane][c] = (__bf16)0.0f;
    }
    __syncthreads();

    // ---- ctx tile = P(16x288) @ V(288x64), 9 K-chunks x 4 N-tiles ------
    for (int nt = 0; nt < 4; ++nt) {
        int n = nt * 16 + ln;
        const __bf16* vrow = Vt + ((size_t)h * DH + n) * S_LEN;  // along seq
        v8f acc = {};
        for (int kc = 0; kc < 9; ++kc) {
            int kb = kc * 32;
            v16bf af = load_frag16(&pr[ln][kb + hf * 8]);   // ds_read_b128 x2
            // B fragment: two aligned 8-element runs along sequence axis.
            int c0 = colStart + kb + hf * 8;
            int c1 = c0 + 16;
            union { u32x4 q[2]; v16bf v; } u;
            u32x4 z = {};
            u.q[0] = (c0 >= 0 && c0 + 8 <= S_LEN) ? *(const u32x4*)(vrow + c0) : z;
            u.q[1] = (c1 >= 0 && c1 + 8 <= S_LEN) ? *(const u32x4*)(vrow + c1) : z;
            acc = wmma_bf16(af, u.v, acc);
        }
#pragma unroll
        for (int r = 0; r < 8; ++r)
            CTX[(size_t)(row0 + r + hf * 8) * HD + h * DH + n] = (__bf16)acc[r];
    }
}

// ---------------------------------------------------------------------------
// Host-side orchestration
// ---------------------------------------------------------------------------
extern "C" void kernel_launch(void* const* d_in, const int* in_sizes, int n_in,
                              void* d_out, int out_size, void* d_ws, size_t ws_size,
                              hipStream_t stream) {
    const float* hs   = (const float*)d_in[0];               // (1,2048,1024)
    const unsigned char* am = (const unsigned char*)d_in[1]; // (1,2048) bool
    const float* Wq   = (const float*)d_in[2];
    const float* Wk   = (const float*)d_in[3];
    const float* Wv   = (const float*)d_in[4];
    const float* pq   = (const float*)d_in[5];               // (1,16,64,257)
    const float* pk   = (const float*)d_in[6];
    const float* Wo   = (const float*)d_in[7];
    const float* bo   = (const float*)d_in[8];
    float* out = (float*)d_out;

    const size_t nX  = (size_t)S_LEN * HD;        // 2M
    const size_t nW  = (size_t)HD * HD;           // 1M
    const size_t nPT = (size_t)NH * BAND * DH;    // 278528
    const size_t nCB = (size_t)NH * S_LEN * BAND; // 8.9M floats

    char* ws = (char*)d_ws;
    size_t off = 0;
    auto carve = [&](size_t bytes) {
        char* p = ws + off;
        off += (bytes + 255) & ~(size_t)255;
        return p;
    };
    __bf16* Xb   = (__bf16*)carve(nX * 2);
    __bf16* Wqb  = (__bf16*)carve(nW * 2);
    __bf16* Wkb  = (__bf16*)carve(nW * 2);
    __bf16* Wvb  = (__bf16*)carve(nW * 2);
    __bf16* Wob  = (__bf16*)carve(nW * 2);
    __bf16* PqT  = (__bf16*)carve(nPT * 2);
    __bf16* PkT  = (__bf16*)carve(nPT * 2);
    __bf16* Qb   = (__bf16*)carve(nX * 2);
    __bf16* Kb   = (__bf16*)carve(nX * 2);
    __bf16* Vb   = (__bf16*)carve(nX * 2);
    __bf16* Vtb  = (__bf16*)carve(nX * 2);
    __bf16* Cxb  = (__bf16*)carve(nX * 2);
    float*  C2P  = (float*)carve(nCB * 4);
    float*  P2C  = (float*)carve(nCB * 4);
    (void)ws_size; (void)n_in; (void)in_sizes; (void)out_size;

    auto cvt = [&](const float* src, __bf16* dst, size_t n) {
        cvt_f32_bf16<<<(unsigned)((n + 255) / 256), 256, 0, stream>>>(src, dst, (int)n);
    };
    cvt(hs, Xb, nX);
    cvt(Wq, Wqb, nW);
    cvt(Wk, Wkb, nW);
    cvt(Wv, Wvb, nW);
    cvt(Wo, Wob, nW);

    // Position tables: transpose + pad to (NH, 272, DH) bf16
    {
        int n = (int)nPT;
        int blocks = (n + 255) / 256;
        transpose_pos<<<blocks, 256, 0, stream>>>(pq, PqT);
        transpose_pos<<<blocks, 256, 0, stream>>>(pk, PkT);
    }

    // QKV projections: (2048x1024) @ (1024x1024)^T, 16x64 tiles per wave
    {
        int waves = (S_LEN / 16) * (HD / 64);              // 2048
        int blocks = (waves * 32 + 255) / 256;             // 256
        gemm_bt4<<<blocks, 256, 0, stream>>>(Xb, Wqb, S_LEN, HD, HD, Qb, nullptr, nullptr);
        gemm_bt4<<<blocks, 256, 0, stream>>>(Xb, Wkb, S_LEN, HD, HD, Kb, nullptr, nullptr);
        gemm_bt4<<<blocks, 256, 0, stream>>>(Xb, Wvb, S_LEN, HD, HD, Vb, nullptr, nullptr);
    }

    // V transpose for the ctx phase
    {
        int blocks = (int)((nX + 255) / 256);
        transpose_v<<<blocks, 256, 0, stream>>>(Vb, Vtb);
    }

    // Position GEMMs: c2p = Q_h @ posK_h ; p2c = K_h @ posQ_h
    {
        int waves = (S_LEN / 16) * (BAND / 16);            // 2176 per head
        dim3 grid((waves * 32 + 255) / 256, NH);
        pos_gemm<<<grid, 256, 0, stream>>>(Qb, PkT, C2P);
        pos_gemm<<<grid, 256, 0, stream>>>(Kb, PqT, P2C);
    }

    // Banded attention
    {
        dim3 grid(S_LEN / 16, NH);
        attn_band<<<grid, 32, 0, stream>>>(Qb, Kb, Vtb, C2P, P2C, am, Cxb);
    }

    // Output projection with bias: f32 out
    {
        int waves = (S_LEN / 16) * (HD / 64);
        int blocks = (waves * 32 + 255) / 256;
        gemm_bt4<<<blocks, 256, 0, stream>>>(Cxb, Wob, S_LEN, HD, HD, nullptr, out, bo);
    }
}
